// TiedTropicalFeatureRecovery_37288906063956
// MI455X (gfx1250) — matrix operand order repaired and checked by reading, not compiled, not run on hardware
//
#include <hip/hip_runtime.h>
#include <hip/hip_bf16.h>

// ---------------------------------------------------------------------------
// TiedTropicalFeatureRecovery on gfx1250 (MI455X, wave32, WMMA)
//   F=1024 features, D=256 model dim, H=8 heads, C=64 cells, B=4096 batch
// Stage 1: reps[f, d] = latent + sigmoid-mixed top-2 codes  (VALU, LDS)
// Stage 2: hidden = x @ reps            (v_wmma_f32_16x16x32_f16, pipelined)
// Stage 3: out = relu(hidden @ reps.T + bias)  (v_wmma, pipelined)
// ---------------------------------------------------------------------------

#define F_DIM 1024
#define D_DIM 256
#define HEADS 8
#define CELLS 64
#define BATCH 4096
#define CODE_SCALE 0.35355339059327373f  // 1/sqrt(8)

typedef __attribute__((ext_vector_type(16))) _Float16 v16h;
typedef __attribute__((ext_vector_type(8)))  _Float16 v8h;
typedef __attribute__((ext_vector_type(8)))  float    v8f;

union AFrag { v16h v; v8h h[2]; };

// Load 8 consecutive fp32 and convert to 8 f16 (two b128 loads + cvt_pk)
__device__ __forceinline__ v8h cvt8_f32(const float* __restrict__ p) {
    const float4 a = *(const float4*)(p);
    const float4 b = *(const float4*)(p + 4);
    v8h r;
    r[0] = (_Float16)a.x; r[1] = (_Float16)a.y;
    r[2] = (_Float16)a.z; r[3] = (_Float16)a.w;
    r[4] = (_Float16)b.x; r[5] = (_Float16)b.y;
    r[6] = (_Float16)b.z; r[7] = (_Float16)b.w;
    return r;
}

// 4 WMMAs sharing one A fragment (16x64 wave strip)
__device__ __forceinline__ void wmma4(const AFrag& a, const v16h b[4], v8f acc[4]) {
#pragma unroll
    for (int t = 0; t < 4; ++t)
        acc[t] = __builtin_amdgcn_wmma_f32_16x16x32_f16(
            false, a.v, false, b[t], (short)0, acc[t], false, false);
}

// ---------------------------------------------------------------------------
// Kernel 1: one block (256 threads) per feature f.
//   lat[d] = proj_w[d, f]  (LDS)
//   wave h: scores[h, c] = max_d(lat[d] + router_w[h,c,d]) + router_b[h,c]
//   threads 0..7: top-2 scan per head, w = sigmoid(margin)
//   all threads (d = tid): reps = lat + CODE_SCALE * sum_h mix(code)
// ---------------------------------------------------------------------------
__global__ void __launch_bounds__(256)
reps_kernel(const float* __restrict__ proj_w,
            const float* __restrict__ router_w,
            const float* __restrict__ router_b,
            const float* __restrict__ code,
            _Float16* __restrict__ reps,     // [F, D] f16
            _Float16* __restrict__ repsT) {  // [D, F] f16
    __shared__ float lat[D_DIM];
    __shared__ float sc[HEADS * CELLS];
    __shared__ int   cwin[HEADS];
    __shared__ int   crun[HEADS];
    __shared__ float wmix[HEADS];

    const int f   = blockIdx.x;
    const int tid = threadIdx.x;

    // latent row (transposed gather of proj_w)
    lat[tid] = proj_w[(size_t)tid * F_DIM + f];
    __syncthreads();

    const int h    = tid >> 5;   // wave id == head
    const int lane = tid & 31;
#pragma unroll
    for (int cc = 0; cc < 2; ++cc) {
        const int c = lane + cc * 32;
        const float* wr = router_w + (size_t)(h * CELLS + c) * D_DIM;
        float m = -3.4e38f;
        for (int d = 0; d < D_DIM; ++d)
            m = fmaxf(m, lat[d] + wr[d]);          // tropical max-plus
        sc[h * CELLS + c] = m + router_b[h * CELLS + c];
    }
    __syncthreads();

    if (tid < HEADS) {
        float b1 = -3.4e38f, b2 = -3.4e38f;
        int   i1 = 0,        i2 = 0;
        for (int c = 0; c < CELLS; ++c) {
            const float v = sc[tid * CELLS + c];
            if (v > b1)      { b2 = b1; i2 = i1; b1 = v; i1 = c; }
            else if (v > b2) { b2 = v;  i2 = c; }
        }
        cwin[tid] = i1;
        crun[tid] = i2;
        wmix[tid] = 1.0f / (1.0f + __expf(-(b1 - b2)));  // sigmoid(margin)
    }
    __syncthreads();

    float s = 0.0f;
#pragma unroll
    for (int hh = 0; hh < HEADS; ++hh) {
        const float w = wmix[hh];
        s += w          * code[(size_t)(hh * CELLS + cwin[hh]) * D_DIM + tid]
           + (1.0f - w) * code[(size_t)(hh * CELLS + crun[hh]) * D_DIM + tid];
    }
    const float rep = lat[tid] + CODE_SCALE * s;
    reps [(size_t)f   * D_DIM + tid] = (_Float16)rep;
    repsT[(size_t)tid * F_DIM + f]   = (_Float16)rep;
}

// ---------------------------------------------------------------------------
// Kernel 2: hidden[B, D] = x[B, F] @ reps[F, D]   (K = 1024)
//   block = 8 waves, 2 (M) x 4 (N): block tile 32 x 256, grid.x = B/32
//   wave tile 16 x 64 -> 4 accumulators; double-buffered K pipeline so the
//   loads for k+32 are in flight while the WMMAs for k execute.
// ---------------------------------------------------------------------------
__global__ void __launch_bounds__(256)
gemm1_kernel(const float* __restrict__ x,
             const _Float16* __restrict__ repsT,  // [D, F]
             _Float16* __restrict__ hidden) {     // [B, D] f16
    const int tid   = threadIdx.x;
    const int lane  = tid & 31;
    const int wid   = tid >> 5;
    const int waveM = wid >> 2;
    const int waveN = wid & 3;

    const int m0    = blockIdx.x * 32 + waveM * 16;
    const int row   = m0 + (lane & 15);
    const int baseA = (lane >> 4) * 16;   // A: halfs 0-7 at K+baseA, 8-15 at K+baseA+8
    const int baseB = (lane >> 4) * 16;   // B: halfs 0-15 at K+baseB (contiguous)
    const int n0    = waveN * 64;

    const float* __restrict__ xrow = x + (size_t)row * F_DIM;
    const _Float16* brow[4];
#pragma unroll
    for (int t = 0; t < 4; ++t) {
        const int d = n0 + 16 * t + (lane & 15);     // output column (model dim)
        brow[t] = repsT + (size_t)d * F_DIM + baseB; // contiguous over K = f
    }

    v8f acc[4] = {};
    AFrag a0, a1;
    v16h  b0[4], b1[4];

    // prologue: k = 0
    a0.h[0] = cvt8_f32(xrow + baseA);
    a0.h[1] = cvt8_f32(xrow + baseA + 8);
#pragma unroll
    for (int t = 0; t < 4; ++t) b0[t] = *(const v16h*)(brow[t]);

    for (int kk = 0; kk < F_DIM - 64; kk += 64) {
        // issue loads for kk+32, then compute kk
        a1.h[0] = cvt8_f32(xrow + kk + 32 + baseA);
        a1.h[1] = cvt8_f32(xrow + kk + 32 + baseA + 8);
#pragma unroll
        for (int t = 0; t < 4; ++t) b1[t] = *(const v16h*)(brow[t] + kk + 32);
        wmma4(a0, b0, acc);

        // issue loads for kk+64, then compute kk+32
        a0.h[0] = cvt8_f32(xrow + kk + 64 + baseA);
        a0.h[1] = cvt8_f32(xrow + kk + 64 + baseA + 8);
#pragma unroll
        for (int t = 0; t < 4; ++t) b0[t] = *(const v16h*)(brow[t] + kk + 64);
        wmma4(a1, b1, acc);
    }
    // epilogue: k = K-64 already in (a0,b0); load K-32 then drain
    a1.h[0] = cvt8_f32(xrow + F_DIM - 32 + baseA);
    a1.h[1] = cvt8_f32(xrow + F_DIM - 32 + baseA + 8);
#pragma unroll
    for (int t = 0; t < 4; ++t) b1[t] = *(const v16h*)(brow[t] + F_DIM - 32);
    wmma4(a0, b0, acc);
    wmma4(a1, b1, acc);

    const int mhi = (lane >> 4) * 8;
#pragma unroll
    for (int t = 0; t < 4; ++t) {
        const int col = n0 + 16 * t + (lane & 15);
#pragma unroll
        for (int r = 0; r < 8; ++r) {
            const int m = m0 + r + mhi;
            hidden[(size_t)m * D_DIM + col] = (_Float16)acc[t][r];
        }
    }
}

// ---------------------------------------------------------------------------
// Kernel 3: out[B, F] = relu(hidden[B, D] @ reps.T + bias)   (K = 256)
//   B operand column n = feature f -> contiguous row of reps[F, D].
//   grid = (B/32, F/256); same double-buffered pipeline; fused bias + relu.
// ---------------------------------------------------------------------------
__global__ void __launch_bounds__(256)
gemm2_kernel(const _Float16* __restrict__ hidden,  // [B, D]
             const _Float16* __restrict__ reps,    // [F, D]
             const float* __restrict__ bias,       // [F]
             float* __restrict__ out) {            // [B, F]
    const int tid   = threadIdx.x;
    const int lane  = tid & 31;
    const int wid   = tid >> 5;
    const int waveM = wid >> 2;
    const int waveN = wid & 3;

    const int m0    = blockIdx.x * 32 + waveM * 16;
    const int row   = m0 + (lane & 15);
    const int baseA = (lane >> 4) * 16;
    const int baseB = (lane >> 4) * 16;
    const int n0    = blockIdx.y * 256 + waveN * 64;

    const _Float16* __restrict__ arow = hidden + (size_t)row * D_DIM;
    const _Float16* brow[4];
#pragma unroll
    for (int t = 0; t < 4; ++t) {
        const int fcol = n0 + 16 * t + (lane & 15);    // output column (feature)
        brow[t] = reps + (size_t)fcol * D_DIM + baseB; // contiguous over K = d
    }

    v8f acc[4] = {};
    AFrag a0, a1;
    v16h  b0[4], b1[4];

    // prologue: k = 0
    a0.h[0] = *(const v8h*)(arow + baseA);
    a0.h[1] = *(const v8h*)(arow + baseA + 8);
#pragma unroll
    for (int t = 0; t < 4; ++t) b0[t] = *(const v16h*)(brow[t]);

    for (int kk = 0; kk < D_DIM - 64; kk += 64) {
        a1.h[0] = *(const v8h*)(arow + kk + 32 + baseA);
        a1.h[1] = *(const v8h*)(arow + kk + 32 + baseA + 8);
#pragma unroll
        for (int t = 0; t < 4; ++t) b1[t] = *(const v16h*)(brow[t] + kk + 32);
        wmma4(a0, b0, acc);

        a0.h[0] = *(const v8h*)(arow + kk + 64 + baseA);
        a0.h[1] = *(const v8h*)(arow + kk + 64 + baseA + 8);
#pragma unroll
        for (int t = 0; t < 4; ++t) b0[t] = *(const v16h*)(brow[t] + kk + 64);
        wmma4(a1, b1, acc);
    }
    a1.h[0] = *(const v8h*)(arow + D_DIM - 32 + baseA);
    a1.h[1] = *(const v8h*)(arow + D_DIM - 32 + baseA + 8);
#pragma unroll
    for (int t = 0; t < 4; ++t) b1[t] = *(const v16h*)(brow[t] + D_DIM - 32);
    wmma4(a0, b0, acc);
    wmma4(a1, b1, acc);

    const int mhi = (lane >> 4) * 8;
#pragma unroll
    for (int t = 0; t < 4; ++t) {
        const int col = n0 + 16 * t + (lane & 15);
        const float b = bias[col];
#pragma unroll
        for (int r = 0; r < 8; ++r) {
            const int m = m0 + r + mhi;
            out[(size_t)m * F_DIM + col] = fmaxf(acc[t][r] + b, 0.0f);
        }
    }
}

// ---------------------------------------------------------------------------
// Host-side launch. Inputs (setup_inputs order):
//   0: x        [4096, 1024] f32
//   1: proj_w   [256, 1024]  f32
//   2: router_w [8, 64, 256] f32
//   3: router_b [8, 64]      f32
//   4: code     [8, 64, 256] f32
//   5: bias     [1024]       f32
// Output: [4096, 1024] f32
// Workspace: reps_f16 (512 KB) | repsT_f16 (512 KB) | hidden_f16 (2 MB)
// ---------------------------------------------------------------------------
extern "C" void kernel_launch(void* const* d_in, const int* in_sizes, int n_in,
                              void* d_out, int out_size, void* d_ws, size_t ws_size,
                              hipStream_t stream) {
    const float* x        = (const float*)d_in[0];
    const float* proj_w   = (const float*)d_in[1];
    const float* router_w = (const float*)d_in[2];
    const float* router_b = (const float*)d_in[3];
    const float* code     = (const float*)d_in[4];
    const float* bias     = (const float*)d_in[5];
    float* out            = (float*)d_out;

    char* wsb = (char*)d_ws;
    _Float16* reps   = (_Float16*)(wsb);                       // F*D f16
    _Float16* repsT  = (_Float16*)(wsb + (size_t)F_DIM * D_DIM * 2);
    _Float16* hidden = (_Float16*)(wsb + (size_t)F_DIM * D_DIM * 4);

    // Stage 1: tied tropical routing -> reps (and transpose)
    reps_kernel<<<F_DIM, 256, 0, stream>>>(proj_w, router_w, router_b, code,
                                           reps, repsT);
    // Stage 2: hidden = x @ reps
    gemm1_kernel<<<BATCH / 32, 256, 0, stream>>>(x, repsT, hidden);
    // Stage 3: out = relu(hidden @ reps.T + bias)
    gemm2_kernel<<<dim3(BATCH / 32, F_DIM / 256), 256, 0, stream>>>(hidden, reps,
                                                                    bias, out);
}
